// MultiheadSelfAttention_80616536146388
// MI455X (gfx1250) — matrix operand order repaired and checked by reading, not compiled, not run on hardware
//
#include <hip/hip_runtime.h>
#include <hip/hip_bf16.h>
#include <math.h>

#define DM 1024
#define NH 16
#define HD 64
#define BB 2
#define TT 2048

typedef __attribute__((ext_vector_type(16))) __bf16           v16bf;
typedef __attribute__((ext_vector_type(16))) unsigned short   v16us;
typedef __attribute__((ext_vector_type(8)))  unsigned short   v8us;
typedef __attribute__((ext_vector_type(8)))  float            v8f;

// f32 -> bf16 round-to-nearest-even (prepass, epilogues, attention P tile)
static __device__ __forceinline__ unsigned short f2bf(float f){
  union { float f; unsigned u; } c; c.f = f;
  unsigned r = c.u + 0x7fffu + ((c.u >> 16) & 1u);
  return (unsigned short)(r >> 16);
}

// A-fragment swizzle within each 32-wide k block:
// memory order [k0..7, k16..23, k8..15, k24..31] so lane-half lh reads k 16lh..16lh+15
// as two 16B pieces (offsets lh*8, lh*8+16).  grp 0,1,2,3 -> 0,2,1,3
static __device__ __forceinline__ int swz32(int c){
  int kb = c & 31, grp = kb >> 3;
  int dg = ((grp & 1) << 1) | (grp >> 1);
  return (c & ~31) | (dg << 3) | (kb & 7);
}

static __device__ __forceinline__ v8f wmma_bf16(v16us a, v16us b, v8f c){
  // 8 args: (neg_a, A, neg_b, B, c_mod, C, reuse_a, reuse_b)
  return __builtin_amdgcn_wmma_f32_16x16x32_bf16(
      false, __builtin_bit_cast(v16bf, a),
      false, __builtin_bit_cast(v16bf, b),
      (short)0, c, false, false);
}

static __device__ __forceinline__ v16us ld16_bf(const unsigned short* p){
  return *reinterpret_cast<const v16us*>(p);          // 32B contiguous
}
static __device__ __forceinline__ v16us ld8x2_bf(const unsigned short* p0,
                                                 const unsigned short* p1){
  v8us lo = *reinterpret_cast<const v8us*>(p0);
  v8us hi = *reinterpret_cast<const v8us*>(p1);
  return __builtin_shufflevector(lo, hi, 0,1,2,3,4,5,6,7,8,9,10,11,12,13,14,15);
}

// ---------------------------------------------------------------------------
// Prepass: f32 -> bf16 (weights = B operands, natural layout)
// ---------------------------------------------------------------------------
__global__ __launch_bounds__(256)
void cvt_bf16_kernel(const float* __restrict__ src, unsigned short* __restrict__ dst, int n){
  int i = blockIdx.x * 256 + threadIdx.x;
  if (i < n) dst[i] = f2bf(src[i]);
}
// Prepass: f32 -> bf16 with per-32-block A-fragment swizzle (x = A operand)
__global__ __launch_bounds__(256)
void cvt_bf16_swz_kernel(const float* __restrict__ src, unsigned short* __restrict__ dst, int n){
  int i = blockIdx.x * 256 + threadIdx.x;
  if (i < n){
    int col = i & (DM - 1);
    dst[(i & ~(DM - 1)) + swz32(col)] = f2bf(src[i]);
  }
}

// ---------------------------------------------------------------------------
// Kernel 1: Y = Xb @ Wb^T for W in {wq,wk,wv}; fused RoPE (+1/8 on Q).
// Register-blocked: one wave computes a 32x64 tile (2 A frags x 4 B frags,
// 8 WMMAs per 32-deep k-step; 12 b128 loads / 8 wmma = 1.5 loads per wmma).
// Q,K stored bf16 [b,h,t,d]; V stored transposed bf16 [b,h,d,t].
// ---------------------------------------------------------------------------
__global__ __launch_bounds__(256)
void qkv_rope_kernel(const unsigned short* __restrict__ xb,
                     const unsigned short* __restrict__ wqb,
                     const unsigned short* __restrict__ wkb,
                     const unsigned short* __restrict__ wvb,
                     unsigned short* __restrict__ Q,
                     unsigned short* __restrict__ K,
                     unsigned short* __restrict__ Vt){
  const int lane = threadIdx.x & 31;
  const int wave = threadIdx.x >> 5;
  const int lm = lane & 15, lh = lane >> 4;
  const int m0 = blockIdx.x * 32;                 // 128 row tiles
  const int n0 = (blockIdx.y * 8 + wave) * 64;    // 16 col tiles
  const int which = blockIdx.z;                   // 0=q 1=k 2=v
  const unsigned short* w = (which == 0) ? wqb : (which == 1) ? wkb : wvb;

  const unsigned short* ar0 = xb + (size_t)(m0 + lm) * DM + lh * 8;        // swizzled A
  const unsigned short* ar1 = ar0 + (size_t)16 * DM;
  const unsigned short* br0 = w + (size_t)(n0 + lm) * DM + lh * 16;        // natural B
  const unsigned short* br1 = br0 + (size_t)16 * DM;
  const unsigned short* br2 = br0 + (size_t)32 * DM;
  const unsigned short* br3 = br0 + (size_t)48 * DM;

  v8f acc[2][4] = {};
#pragma unroll 2
  for (int k0 = 0; k0 < DM; k0 += 32){
    __builtin_prefetch(ar0 + k0 + 256, 0, 3);     // global_prefetch_b8
    __builtin_prefetch(ar1 + k0 + 256, 0, 3);
    v16us a0 = ld8x2_bf(ar0 + k0, ar0 + k0 + 16);
    v16us a1 = ld8x2_bf(ar1 + k0, ar1 + k0 + 16);
    v16us b0 = ld16_bf(br0 + k0);
    v16us b1 = ld16_bf(br1 + k0);
    v16us b2 = ld16_bf(br2 + k0);
    v16us b3 = ld16_bf(br3 + k0);
    acc[0][0] = wmma_bf16(a0, b0, acc[0][0]);
    acc[0][1] = wmma_bf16(a0, b1, acc[0][1]);
    acc[0][2] = wmma_bf16(a0, b2, acc[0][2]);
    acc[0][3] = wmma_bf16(a0, b3, acc[0][3]);
    acc[1][0] = wmma_bf16(a1, b0, acc[1][0]);
    acc[1][1] = wmma_bf16(a1, b1, acc[1][1]);
    acc[1][2] = wmma_bf16(a1, b2, acc[1][2]);
    acc[1][3] = wmma_bf16(a1, b3, acc[1][3]);
  }

  const int h = n0 >> 6;                          // head fixed per wave (n0 % 64 == 0)
  if (which == 2){
#pragma unroll
    for (int j = 0; j < 4; ++j){
      const int d = j * 16 + lm;
#pragma unroll
      for (int i = 0; i < 2; ++i){
#pragma unroll
        for (int v = 0; v < 8; ++v){
          const int row = m0 + i * 16 + v + 8 * lh;
          const int t = row & (TT - 1), b_ = row >> 11;
          Vt[((size_t)(b_ * NH + h) * HD + d) * TT + t] = f2bf(acc[i][j][v]);
        }
      }
    }
  } else {
    unsigned short* dst = (which == 0) ? Q : K;
    const float scale = (which == 0) ? 0.125f : 1.0f;   // fold 1/sqrt(64) into Q
#pragma unroll
    for (int j = 0; j < 4; ++j){
      const int d = j * 16 + lm;
      const int jj = d >> 1;
      const float inv = __expf(-(float)(2 * jj) * (9.210340371976184f / 64.0f)); // theta^(-2jj/64)
#pragma unroll
      for (int i = 0; i < 2; ++i){
#pragma unroll
        for (int v = 0; v < 8; ++v){
          float val = acc[i][j][v];
          float pv  = __shfl_xor(val, 1, 32);     // rotation partner (adjacent lane)
          const int row = m0 + i * 16 + v + 8 * lh;
          const int t = row & (TT - 1), b_ = row >> 11;
          float s, c;
          __sincosf((float)t * inv, &s, &c);
          float out = (d & 1) ? (pv * s + val * c) : (val * c - pv * s);
          out *= scale;
          dst[((size_t)(b_ * NH + h) * TT + t) * HD + d] = f2bf(out);
        }
      }
    }
  }
}

// ---------------------------------------------------------------------------
// Kernel 2: causal flash attention. One wave per (b,h,q-tile of 16 rows).
// 32-key steps: 4 WMMA scores, online softmax via shfl_xor in 16-lane halves,
// P D->A transpose through per-wave LDS, 4 WMMA accumulate P@V (16x64 f32).
// Output written as swizzled bf16 so oproj's split A loads give k = 16lh + s.
// ---------------------------------------------------------------------------
__global__ __launch_bounds__(256)
void attn_kernel(const unsigned short* __restrict__ Q,
                 const unsigned short* __restrict__ K,
                 const unsigned short* __restrict__ Vt,
                 unsigned short* __restrict__ aoutb){
  __shared__ __align__(16) unsigned short pbuf[8][16][32];   // per-wave P tile
  const int lane = threadIdx.x & 31;
  const int wave = threadIdx.x >> 5;
  const int lm = lane & 15, lh = lane >> 4;
  const int bh = blockIdx.x;                      // 0..31  (b*16+h)
  const int qt = blockIdx.y * 8 + wave;           // 0..127
  const int b_ = bh >> 4, h = bh & 15;
  const int q0 = qt * 16;

  const unsigned short* Qb = Q  + (size_t)bh * TT * HD;
  const unsigned short* Kb = K  + (size_t)bh * TT * HD;
  const unsigned short* Vb = Vt + (size_t)bh * HD * TT;

  const unsigned short* qrow = Qb + (size_t)(q0 + lm) * HD + lh * 16;
  v16us fq0 = ld16_bf(qrow);                      // d = 0..31
  v16us fq1 = ld16_bf(qrow + 32);                 // d = 32..63

  v8f o0 = {}, o1 = {}, o2 = {}, o3 = {};
  float mrow[8], lrow[8];
#pragma unroll
  for (int v = 0; v < 8; ++v){ mrow[v] = -__builtin_inff(); lrow[v] = 0.0f; }

  for (int kb = 0; kb < q0 + 16; kb += 32){
    const unsigned short* k0r = Kb + (size_t)(kb + lm) * HD + lh * 16;
    const unsigned short* k1r = k0r + 16 * HD;
    v8f s0 = {}, s1 = {};
    s0 = wmma_bf16(fq0, ld16_bf(k0r),      s0);
    s0 = wmma_bf16(fq1, ld16_bf(k0r + 32), s0);
    s1 = wmma_bf16(fq0, ld16_bf(k1r),      s1);
    s1 = wmma_bf16(fq1, ld16_bf(k1r + 32), s1);

    const int key0 = kb + lm, key1 = kb + 16 + lm;
#pragma unroll
    for (int v = 0; v < 8; ++v){
      const int qr = q0 + v + 8 * lh;
      float a = (key0 <= qr) ? s0[v] : -__builtin_inff();
      float b = (key1 <= qr) ? s1[v] : -__builtin_inff();
      float t = fmaxf(a, b);                      // row max across 16 lanes
      t = fmaxf(t, __shfl_xor(t, 1, 32));
      t = fmaxf(t, __shfl_xor(t, 2, 32));
      t = fmaxf(t, __shfl_xor(t, 4, 32));
      t = fmaxf(t, __shfl_xor(t, 8, 32));
      float mnew  = fmaxf(mrow[v], t);
      float alpha = __expf(mrow[v] - mnew);
      mrow[v] = mnew;
      float p0 = __expf(a - mnew);
      float p1 = __expf(b - mnew);
      float rs = p0 + p1;                         // row sum across 16 lanes
      rs += __shfl_xor(rs, 1, 32);
      rs += __shfl_xor(rs, 2, 32);
      rs += __shfl_xor(rs, 4, 32);
      rs += __shfl_xor(rs, 8, 32);
      lrow[v] = lrow[v] * alpha + rs;
      o0[v] *= alpha; o1[v] *= alpha; o2[v] *= alpha; o3[v] *= alpha;
      const int m = v + 8 * lh;                   // D-layout -> [m][k] in LDS
      pbuf[wave][m][lm]      = f2bf(p0);
      pbuf[wave][m][lm + 16] = f2bf(p1);
    }
    // reload P in A-fragment layout (same wave, LDS is in-order)
    const unsigned short* pr = &pbuf[wave][lm][lh * 16];
    v16us fp = ld8x2_bf(pr, pr + 8);
    const unsigned short* v0r = Vb + (size_t)lm * TT + kb + lh * 16;
    o0 = wmma_bf16(fp, ld16_bf(v0r),           o0);
    o1 = wmma_bf16(fp, ld16_bf(v0r + 16 * TT), o1);
    o2 = wmma_bf16(fp, ld16_bf(v0r + 32 * TT), o2);
    o3 = wmma_bf16(fp, ld16_bf(v0r + 48 * TT), o3);
  }

  unsigned short* orow = aoutb + (size_t)b_ * TT * DM;
  // swizzled column positions (A-fragment layout for the output projection)
  const int c0 = swz32(h * HD + 0  + lm);
  const int c1 = swz32(h * HD + 16 + lm);
  const int c2 = swz32(h * HD + 32 + lm);
  const int c3 = swz32(h * HD + 48 + lm);
#pragma unroll
  for (int v = 0; v < 8; ++v){
    const int m = v + 8 * lh;
    const float invl = 1.0f / lrow[v];
    unsigned short* p = orow + (size_t)(q0 + m) * DM;
    p[c0] = f2bf(o0[v] * invl);
    p[c1] = f2bf(o1[v] * invl);
    p[c2] = f2bf(o2[v] * invl);
    p[c3] = f2bf(o3[v] * invl);
  }
}

// ---------------------------------------------------------------------------
// Kernel 3: out = attn_out @ wo^T (bf16 operands, f32 accumulate/output).
// Same 32x64 register blocking as kernel 1.
// ---------------------------------------------------------------------------
__global__ __launch_bounds__(256)
void oproj_kernel(const unsigned short* __restrict__ aoutb,
                  const unsigned short* __restrict__ wob,
                  float* __restrict__ out){
  const int lane = threadIdx.x & 31;
  const int wave = threadIdx.x >> 5;
  const int lm = lane & 15, lh = lane >> 4;
  const int m0 = blockIdx.x * 32;
  const int n0 = (blockIdx.y * 8 + wave) * 64;

  const unsigned short* ar0 = aoutb + (size_t)(m0 + lm) * DM + lh * 8;     // swizzled A
  const unsigned short* ar1 = ar0 + (size_t)16 * DM;
  const unsigned short* br0 = wob + (size_t)(n0 + lm) * DM + lh * 16;      // natural B
  const unsigned short* br1 = br0 + (size_t)16 * DM;
  const unsigned short* br2 = br0 + (size_t)32 * DM;
  const unsigned short* br3 = br0 + (size_t)48 * DM;

  v8f acc[2][4] = {};
#pragma unroll 2
  for (int k0 = 0; k0 < DM; k0 += 32){
    __builtin_prefetch(ar0 + k0 + 256, 0, 3);
    __builtin_prefetch(ar1 + k0 + 256, 0, 3);
    v16us a0 = ld8x2_bf(ar0 + k0, ar0 + k0 + 16);
    v16us a1 = ld8x2_bf(ar1 + k0, ar1 + k0 + 16);
    v16us b0 = ld16_bf(br0 + k0);
    v16us b1 = ld16_bf(br1 + k0);
    v16us b2 = ld16_bf(br2 + k0);
    v16us b3 = ld16_bf(br3 + k0);
    acc[0][0] = wmma_bf16(a0, b0, acc[0][0]);
    acc[0][1] = wmma_bf16(a0, b1, acc[0][1]);
    acc[0][2] = wmma_bf16(a0, b2, acc[0][2]);
    acc[0][3] = wmma_bf16(a0, b3, acc[0][3]);
    acc[1][0] = wmma_bf16(a1, b0, acc[1][0]);
    acc[1][1] = wmma_bf16(a1, b1, acc[1][1]);
    acc[1][2] = wmma_bf16(a1, b2, acc[1][2]);
    acc[1][3] = wmma_bf16(a1, b3, acc[1][3]);
  }
#pragma unroll
  for (int j = 0; j < 4; ++j){
#pragma unroll
    for (int i = 0; i < 2; ++i){
#pragma unroll
      for (int v = 0; v < 8; ++v){
        const int m = m0 + i * 16 + v + 8 * lh;
        out[(size_t)m * DM + n0 + j * 16 + lm] = acc[i][j][v];
      }
    }
  }
}

// ---------------------------------------------------------------------------
// ws layout (48 MB):
//   xb   bf16 swz   @  0 MB (8 MB)
//   wqb/wkb/wvb/wob @ 8/10/12/14 MB (2 MB each)
//   Q / K / Vt bf16 @ 16/24/32 MB (8 MB each)
//   aout bf16 swz   @ 40 MB (8 MB)
// ---------------------------------------------------------------------------
extern "C" void kernel_launch(void* const* d_in, const int* in_sizes, int n_in,
                              void* d_out, int out_size, void* d_ws, size_t ws_size,
                              hipStream_t stream){
  const float* x  = (const float*)d_in[0];
  const float* wq = (const float*)d_in[1];
  const float* wk = (const float*)d_in[2];
  const float* wv = (const float*)d_in[3];
  const float* wo = (const float*)d_in[4];
  (void)in_sizes; (void)n_in; (void)out_size; (void)ws_size; // token_pos (d_in[5]) unused by reference math

  char* ws = (char*)d_ws;
  const size_t MB = 1024 * 1024;
  unsigned short* xb  = (unsigned short*)(ws);
  unsigned short* wqb = (unsigned short*)(ws + 8  * MB);
  unsigned short* wkb = (unsigned short*)(ws + 10 * MB);
  unsigned short* wvb = (unsigned short*)(ws + 12 * MB);
  unsigned short* wob = (unsigned short*)(ws + 14 * MB);
  unsigned short* Q   = (unsigned short*)(ws + 16 * MB);
  unsigned short* K   = (unsigned short*)(ws + 24 * MB);
  unsigned short* Vt  = (unsigned short*)(ws + 32 * MB);
  unsigned short* ao  = (unsigned short*)(ws + 40 * MB);

  const int NX = BB * TT * DM;   // 4,194,304
  const int NW = DM * DM;        // 1,048,576
  cvt_bf16_swz_kernel<<<dim3(NX / 256), 256, 0, stream>>>(x,  xb,  NX);
  cvt_bf16_kernel    <<<dim3(NW / 256), 256, 0, stream>>>(wq, wqb, NW);
  cvt_bf16_kernel    <<<dim3(NW / 256), 256, 0, stream>>>(wk, wkb, NW);
  cvt_bf16_kernel    <<<dim3(NW / 256), 256, 0, stream>>>(wv, wvb, NW);
  cvt_bf16_kernel    <<<dim3(NW / 256), 256, 0, stream>>>(wo, wob, NW);

  qkv_rope_kernel<<<dim3(128, 2, 3), 256, 0, stream>>>(xb, wqb, wkb, wvb, Q, K, Vt);
  attn_kernel   <<<dim3(32, 16),     256, 0, stream>>>(Q, K, Vt, ao);
  oproj_kernel  <<<dim3(128, 2),     256, 0, stream>>>(ao, wob, (float*)d_out);
}
